// Decoder_39350490366168
// MI455X (gfx1250) — matrix operand order repaired and checked by reading, not compiled, not run on hardware
//
#include <hip/hip_runtime.h>
#include <hip/hip_bf16.h>

typedef __attribute__((ext_vector_type(16))) _Float16 v16h;
typedef __attribute__((ext_vector_type(8)))  _Float16 v8h;
typedef __attribute__((ext_vector_type(8)))  float    v8f;
typedef __attribute__((ext_vector_type(4)))  float    v4f;

#define T_STEPS 32
#define NBATCH  4

// ---------------------------------------------------------------------------
// zero the 3 per-block spike-count accumulators (graph replay safe)
__global__ void init_sums_kernel(float* sums) {
    if (threadIdx.x < 3) sums[threadIdx.x] = 0.0f;
}

// ---------------------------------------------------------------------------
// Pack weights in WMMA B-fragment-linear order:
//   wf[tap][chunk][nt][lane][e] = 2 * w[cin][cout][ky][kx]   (f16, zero-padded)
// where k_local = 8*(lane/16)+(e%8)+16*(e/8), cin = chunk*32+k_local, cout = nt*16+(lane%16).
// A B-fragment is then one contiguous 32B run per lane -> 2 coalesced b128 loads.
__global__ void pack_weights_kernel(const float* __restrict__ w,
                                    _Float16* __restrict__ wf,
                                    int CIN, int COUT, int COUT_PAD) {
    const int NC = CIN / 32, NT = COUT_PAD / 16;
    int idx = blockIdx.x * 256 + threadIdx.x;
    int total = 9 * NC * NT * 512;
    if (idx >= total) return;
    int e    = idx & 15;
    int lane = (idx >> 4) & 31;
    int g    = idx >> 9;              // (tap, chunk, nt) flattened
    int nt   = g % NT;
    int chunk = (g / NT) % NC;
    int tap  = g / (NT * NC);         // ky*3+kx
    int k_local = ((lane >> 4) << 3) + (e & 7) + ((e >> 3) << 4);
    int cin  = chunk * 32 + k_local;
    int cout = nt * 16 + (lane & 15);
    float v = 0.0f;
    if (cout < COUT) v = 2.0f * w[(cin * COUT + cout) * 9 + tap];
    wf[idx] = (_Float16)v;
}

// ---------------------------------------------------------------------------
// One-time transpose of block-1 input to channel-last f16:
//   x [4][128][256][32] f32  ->  xT [4][256][32][128] f16   (n, pix, t, cin)
__global__ void transpose_in_kernel(const float* __restrict__ x,
                                    _Float16* __restrict__ xT) {
    int idx = blockIdx.x * 256 + threadIdx.x;     // 4*128*256*32 = 4,194,304
    if (idx >= 4 * 128 * 256 * 32) return;
    int t   = idx & 31;
    int cin = (idx >> 5) & 127;
    int p   = (idx >> 12) & 255;
    int n   = idx >> 20;
    xT[((n * 256 + p) * 32 + t) * 128 + cin] =
        (_Float16)x[((n * 128 + cin) * 256 + p) * 32 + t];
}

// ---------------------------------------------------------------------------
// One workgroup per output pixel; 8 wave32s, wave w owns M-tile w (M=(n,t)).
// A and B fragments load straight from global (channel-last spikes + packed
// weights), no staging LDS; LDS holds only the Z tile for the fused LIF scan.
template <int CIN, int COUT_REAL, int COUT_PAD, int HIN, int WIN>
__global__ __launch_bounds__(256) void cuba_block_kernel(
    const _Float16* __restrict__ xT,      // [N][HIN][WIN][T][CIN] f16
    const _Float16* __restrict__ wf,      // [9][CIN/32][NT][32][16] f16 (x2 scale)
    _Float16* __restrict__ spkT,          // [N][HO][WO][T][COUT_REAL] f16 or null
    float* __restrict__ f32_out,          // [N][COUT_REAL][HO][WO][T] f32 or null
    float* __restrict__ sum_accum)        // spike-count accumulator (1 float)
{
    constexpr int HO = 2 * HIN - 1, WO = 2 * WIN - 1;
    constexpr int NT = COUT_PAD / 16;     // N-tiles
    constexpr int NC = CIN / 32;          // K-chunks per tap

    __shared__ __align__(16) float Z_lds[128 * COUT_PAD];

    const int tid  = threadIdx.x;
    const int lane = tid & 31;
    const int wid  = tid >> 5;            // M-tile, 0..7
    const int hi   = lane >> 4;           // K-half select
    const int lr   = lane & 15;           // M (A/D) / N (B/D) within tile

    const int m  = wid * 16 + lr;         // this lane's GEMM row
    const int an = m >> 5, at = m & 31;   // (n, t)

    const int pix = blockIdx.x;
    const int oy = pix / WO, ox = pix % WO;

    // Enumerate transpose-conv taps: oy = 2*iy - 1 + ky  (uniform per block)
    int iys[2], kys[2], nky = 0;
    for (int ky = 0; ky < 3; ++ky) {
        int ty = oy + 1 - ky;
        if (ty & 1) continue;
        int iy = ty >> 1;
        if (iy < 0 || iy >= HIN) continue;
        iys[nky] = iy; kys[nky] = ky; ++nky;
    }
    int ixs[2], kxs[2], nkx = 0;
    for (int kx = 0; kx < 3; ++kx) {
        int tx = ox + 1 - kx;
        if (tx & 1) continue;
        int ix = tx >> 1;
        if (ix < 0 || ix >= WIN) continue;
        ixs[nkx] = ix; kxs[nkx] = kx; ++nkx;
    }

    v8f acc[NT];
#pragma unroll
    for (int i = 0; i < NT; ++i) acc[i] = (v8f){};

    for (int a = 0; a < nky; ++a) {
        for (int b = 0; b < nkx; ++b) {
            const int iy = iys[a], ix = ixs[b];
            const int widx = kys[a] * 3 + kxs[b];
            // A: lane-private base for this (n, iy, ix, t); K-runs contiguous
            const _Float16* abase =
                xT + ((an * (HIN * WIN) + iy * WIN + ix) * T_STEPS + at) * CIN + hi * 8;
            const _Float16* bbase = wf + (widx * NC * NT) * 512 + lane * 16;
#pragma unroll
            for (int c = 0; c < NC; ++c) {
                if (c + 1 < NC)               // global_prefetch_b8 next K-chunk
                    __builtin_prefetch(abase + (c + 1) * 32, 0, 0);
                union { v16h v; v8h h[2]; } af;
                af.h[0] = *(const v8h*)(abase + c * 32);       // K {0..7}/{8..15}
                af.h[1] = *(const v8h*)(abase + c * 32 + 16);  // K +16
#pragma unroll
                for (int nt = 0; nt < NT; ++nt) {
                    union { v16h v; v8h h[2]; } bf;
                    const _Float16* bp = bbase + (c * NT + nt) * 512;
                    bf.h[0] = *(const v8h*)(bp);
                    bf.h[1] = *(const v8h*)(bp + 8);
                    acc[nt] = __builtin_amdgcn_wmma_f32_16x16x32_f16(
                        false, af.v, false, bf.v, (short)0, acc[nt], false, false);
                }
            }
        }
    }

    // ---- D -> LDS: lane l VGPR r holds (M = r + 8*(l/16), N = l%16)
#pragma unroll
    for (int nt = 0; nt < NT; ++nt) {
#pragma unroll
        for (int r = 0; r < 8; ++r) {
            Z_lds[(wid * 16 + r + 8 * hi) * COUT_PAD + nt * 16 + lr] = acc[nt][r];
        }
    }
    __syncthreads();

    // ---- fused CUBA LIF scan: one thread per (n, cout)
    int cnt = 0;
    const int scanN = tid / COUT_PAD;
    const int cout  = tid % COUT_PAD;
    const bool active = (scanN < NBATCH) && (cout < COUT_REAL);
    if (active) {
        float cur = 0.0f, vol = 0.0f;
        v4f fov[8];
#pragma unroll
        for (int t = 0; t < T_STEPS; ++t) {
            float z = Z_lds[(scanN * 32 + t) * COUT_PAD + cout];
            cur = 0.5f * cur + z;            // (1 - CURRENT_DECAY)*cur + z
            vol = 0.5f * vol + cur;          // (1 - VOLTAGE_DECAY)*vol + cur
            bool fire = (vol >= 1.0f);
            float s = fire ? 1.0f : 0.0f;
            vol = fire ? 0.0f : vol;         // reset on spike
            cnt += fire;
            fov[t >> 2][t & 3] = s;
            if (spkT)                         // channel-last for next block's A
                spkT[((scanN * (HO * WO) + pix) * T_STEPS + t) * COUT_REAL + cout] =
                    (_Float16)s;
        }
        if (f32_out) {                        // reference layout [N][C][HO][WO][T]
            v4f* p = (v4f*)(f32_out +
                     (((long)scanN * COUT_REAL + cout) * (HO * WO) + pix) * T_STEPS);
#pragma unroll
            for (int j = 0; j < 8; ++j) p[j] = fov[j];
        }
    }
    // wave32 reduction of spike counts, one atomic per wave
    for (int off = 16; off > 0; off >>= 1) cnt += __shfl_down(cnt, off);
    if (lane == 0) atomicAdd(sum_accum, (float)cnt);
}

// ---------------------------------------------------------------------------
__global__ void finalize_counts_kernel(const float* __restrict__ sums,
                                       float* __restrict__ out) {
    int i = threadIdx.x;
    if (i < 3) {
        const float denom[3] = {7872512.0f, 15241216.0f, 3748096.0f};
        out[i] = sums[i] / denom[i];
    }
}

// ---------------------------------------------------------------------------
extern "C" void kernel_launch(void* const* d_in, const int* in_sizes, int n_in,
                              void* d_out, int out_size, void* d_ws, size_t ws_size,
                              hipStream_t stream) {
    (void)in_sizes; (void)n_in; (void)out_size; (void)ws_size;
    const float* x  = (const float*)d_in[0];   // [4,128,16,16,32]
    const float* w1 = (const float*)d_in[1];   // [128,64,3,3]
    const float* w2 = (const float*)d_in[2];   // [64,32,3,3]
    const float* w3 = (const float*)d_in[3];   // [32,2,3,3]
    float* out = (float*)d_out;                // 3,748,096 spikes + 3 counts

    char* ws = (char*)d_ws;                    // 256B-aligned carve-outs
    float*    sums = (float*)(ws + 0);
    _Float16* wf1  = (_Float16*)(ws + 256);        // 9*4*4*512 h = 147456 B
    _Float16* wf2  = (_Float16*)(ws + 147712);     // 9*2*2*512 h = 36864 B
    _Float16* wf3  = (_Float16*)(ws + 184576);     // 9*1*1*512 h = 9216 B
    _Float16* xT1  = (_Float16*)(ws + 193792);     // 4,194,304 h = 8,388,608 B
    _Float16* spk1 = (_Float16*)(ws + 8582400);    // 7,872,512 h = 15,745,024 B
    _Float16* spk2 = (_Float16*)(ws + 24327424);   // 15,241,216 h = 30,482,432 B

    init_sums_kernel<<<1, 32, 0, stream>>>(sums);
    pack_weights_kernel<<<(9 * 4 * 4 * 512 + 255) / 256, 256, 0, stream>>>(w1, wf1, 128, 64, 64);
    pack_weights_kernel<<<(9 * 2 * 2 * 512 + 255) / 256, 256, 0, stream>>>(w2, wf2, 64, 32, 32);
    pack_weights_kernel<<<(9 * 1 * 1 * 512 + 255) / 256, 256, 0, stream>>>(w3, wf3, 32, 2, 16);
    transpose_in_kernel<<<(4 * 128 * 256 * 32) / 256, 256, 0, stream>>>(x, xT1);

    // block 1: xT1 [4,16,16,32,128] -> spk1 [4,31,31,32,64] (channel-last f16)
    cuba_block_kernel<128, 64, 64, 16, 16>
        <<<31 * 31, 256, 0, stream>>>(xT1, wf1, spk1, nullptr, sums + 0);
    // block 2: -> spk2 [4,61,61,32,32]
    cuba_block_kernel<64, 32, 32, 31, 31>
        <<<61 * 61, 256, 0, stream>>>(spk1, wf2, spk2, nullptr, sums + 1);
    // block 3: -> d_out [4,2,121,121,32] f32 (reference layout)
    cuba_block_kernel<32, 2, 16, 61, 61>
        <<<121 * 121, 256, 0, stream>>>(spk2, wf3, nullptr, out, sums + 2);

    finalize_counts_kernel<<<1, 32, 0, stream>>>(sums, out + 3748096);
}
